// Attention_Layer_12463995094037
// MI455X (gfx1250) — matrix operand order repaired and checked by reading, not compiled, not run on hardware
//
#include <hip/hip_runtime.h>
#include <hip/hip_bf16.h>
#include <stdint.h>

// Sizes from the reference
//   B=8, SH=32, SV=32, DH=256, DV=4096, D=256
// d_in order: h(8,32,256) v(8,32,4096) W(4096,256) U(256,256) b(256) w(256,4096)
// d_out: u(8,32,4096) fp32

typedef __attribute__((ext_vector_type(16))) __bf16 v16bf;
typedef __attribute__((ext_vector_type(8)))  float  v8f;

union BF16Frag {
    v16bf        v;
    uint4        q[2];
    unsigned int u[8];
};

__device__ __forceinline__ unsigned int pack_bf16_rne(float a, float b) {
    unsigned int ua = __float_as_uint(a);
    unsigned int ub = __float_as_uint(b);
    ua = (ua + 0x7FFFu + ((ua >> 16) & 1u)) >> 16;
    ub = (ub + 0x7FFFu + ((ub >> 16) & 1u)) >> 16;
    return (ua & 0xFFFFu) | (ub << 16);
}

__device__ __forceinline__ float tanh_fast(float x) {
    // tanh(x) = (e^{2x}-1)/(e^{2x}+1); inputs here are O(1) (sums of small dots)
    float e = __expf(2.0f * x);
    return (e - 1.0f) / (e + 1.0f);
}

__device__ __forceinline__ v8f wmma_bf16(const v16bf& a, const v16bf& b, const v8f& c) {
    return __builtin_amdgcn_wmma_f32_16x16x32_bf16(false, a, false, b, (short)0, c,
                                                   false, false);
}

// ---------------------------------------------------------------------------
// One-shot fp32 -> bf16 conversion (pair-packed). Memory bound; each GEMM
// operand is converted exactly once instead of per tile-reuse.
// ---------------------------------------------------------------------------
__global__ void cvt_f32_to_bf16(const float* __restrict__ in,
                                unsigned int* __restrict__ out, int npairs) {
    int i = blockIdx.x * blockDim.x + threadIdx.x;
    if (i < npairs) {
        float2 f = ((const float2*)in)[i];
        out[i] = pack_bf16_rne(f.x, f.y);
    }
}

// ---------------------------------------------------------------------------
// Prep GEMM: out(M,N) fp32 = A(M,K) bf16 * Bm(K,N) bf16 [+ bias].
// A frags are b128 loads, B frags 16-bit gathers; all 4 B frags loaded before the WMMA group
// so the scheduler can overlap loads with the matrix pipe (partial loadcnt
// waits instead of loadcnt==0 per WMMA). Pointers advance by constant stride
// (no per-iteration 64-bit multiply).
// ---------------------------------------------------------------------------
__global__ __launch_bounds__(256)
void gemm_bf16_wmma(const unsigned short* __restrict__ A,
                    const unsigned short* __restrict__ Bm,
                    const float* __restrict__ bias, float* __restrict__ out,
                    int M, int N, int K) {
    const int tid  = threadIdx.x;
    const int wave = tid >> 5;
    const int lane = tid & 31;
    const int lo   = lane & 15;
    const int sel  = lane >> 4;            // 0: low half-wave, 1: high
    const int ntiles = N >> 6;
    const int gid  = blockIdx.x * 8 + wave;      // wave-uniform
    const int mt   = gid / ntiles;
    const int nbase = (gid % ntiles) << 6;
    if (mt * 16 >= M) return;

    v8f acc[4] = {};
    // A fragment per ISA 16-bit A layout:
    //   VGPR0-3 = K kb+sel*8 .. +7, VGPR4-7 = K kb+16+sel*8 .. +7
    const unsigned short* arow = A + (size_t)(mt * 16 + lo) * K + sel * 8;
    // B fragment: lane = N column (lo), elements = K in A's table order
    const unsigned short* brow = Bm + (size_t)(sel * 8) * N + nbase + lo;
    const size_t bstep = (size_t)32 * N;

    for (int kb = 0; kb < K; kb += 32) {
        BF16Frag af;
        af.q[0] = *(const uint4*)(arow);
        af.q[1] = *(const uint4*)(arow + 16);
        arow += 32;

        BF16Frag bf[4];
#pragma unroll
        for (int nt = 0; nt < 4; ++nt) {
            for (int j = 0; j < 8; ++j) {
                const unsigned short* p = brow + (size_t)(j < 4 ? 2 * j : 8 + 2 * j) * N + nt * 16;
                bf[nt].u[j] = (unsigned int)p[0] | ((unsigned int)p[N] << 16);
            }
        }
        if (kb + 32 < K) __builtin_prefetch(brow + bstep, 0, 1);
        brow += bstep;
#pragma unroll
        for (int nt = 0; nt < 4; ++nt)
            acc[nt] = wmma_bf16(af.v, bf[nt].v, acc[nt]);
    }

#pragma unroll
    for (int nt = 0; nt < 4; ++nt) {
        int col = nbase + nt * 16 + lo;
        float bb = bias ? bias[col] : 0.0f;
#pragma unroll
        for (int r = 0; r < 8; ++r) {
            int row = mt * 16 + r + sel * 8;   // C layout: VGPR r -> M=r / M=8+r
            out[(size_t)row * N + col] = acc[nt][r] + bb;
        }
    }
}

// ---------------------------------------------------------------------------
// Fused main kernel. Block = (h, 512-col group), 8 waves, wave = batch b.
//   Phase 1: f = tanh(Wv[b,v,:] + Uh[b,h,:]) -> bf16 in LDS (256 x 264 rows)
//   Phase 2: per 64-col tile: q = f @ w_bf16 (32x64x256 WMMA per wave)
//   Phase 3: e=exp(q); denom = ds_add_f32 across 8 waves; u = sum_v beta*v
// q/beta (2 x 134 MB) never touch memory.
// ---------------------------------------------------------------------------
static constexpr int ASTR = 264;  // bf16 elems/row: 256 + 8 pad -> 132-bank row
                                  // stride, conflict-free b128 A-frag fetch

__global__ __launch_bounds__(256)
void fused_attention(const float* __restrict__ vin,       // (8,32,4096)
                     const float* __restrict__ Uh,        // (8,32,256) bias folded
                     const float* __restrict__ Wv,        // (8,32,256)
                     const unsigned short* __restrict__ wbf,  // (256,4096) bf16
                     float* __restrict__ out) {           // (8,32,4096)
    const int h   = blockIdx.x;   // 0..31
    const int cg  = blockIdx.y;   // 0..7  (512 columns each)
    const int tid = threadIdx.x;
    const int b   = tid >> 5;     // wave id == batch index
    const int lane = tid & 31;
    const int lo  = lane & 15;
    const int sel = lane >> 4;

    extern __shared__ char smem[];
    unsigned short* Alds = (unsigned short*)smem;          // 256*264*2 = 132 KB
    float* S = (float*)(smem + 256 * ASTR * 2);            // 32*64*4   =   8 KB

    // ---- Phase 1: build f tile for this wave's batch (rows b*32+v) ----
    const float2* uh2 = (const float2*)(Uh + (size_t)(b * 32 + h) * 256);
    for (int vv = 0; vv < 32; ++vv) {
        const float2* wv2 = (const float2*)(Wv + (size_t)(b * 32 + vv) * 256);
        unsigned short* drow = Alds + (b * 32 + vv) * ASTR;
#pragma unroll
        for (int j = 0; j < 4; ++j) {
            int p = lane + 32 * j;             // float2 pair index, 0..127
            float2 a = wv2[p];
            float2 c = uh2[p];
            float t0 = tanh_fast(a.x + c.x);
            float t1 = tanh_fast(a.y + c.y);
            *(unsigned int*)(&drow[2 * p]) = pack_bf16_rne(t0, t1);
        }
    }
    __syncthreads();

    // ---- Phases 2+3 per 64-column tile ----
    for (int ct = 0; ct < 8; ++ct) {
        const int cbase = cg * 512 + ct * 64;

        v8f acc[8] = {};   // [nt] rows v0..15, [4+nt] rows v16..31
        const char* ap0 = (const char*)Alds +
                          (size_t)(b * 32 + lo) * (ASTR * 2) + sel * 16;
        const unsigned short* bro = wbf + (size_t)(sel * 8) * 4096 + cbase + lo;
        const size_t bstep = (size_t)32 * 4096;

        for (int kb = 0; kb < 256; kb += 32) {
            // A frags from LDS (row-major bf16, ISA A layout -> two b128 each)
            BF16Frag a0, a1;
            a0.q[0] = *(const uint4*)(ap0);
            a0.q[1] = *(const uint4*)(ap0 + 32);
            const char* ap1 = ap0 + 16 * ASTR * 2;
            a1.q[0] = *(const uint4*)(ap1);
            a1.q[1] = *(const uint4*)(ap1 + 32);
            ap0 += 64;                         // 32 bf16 K-values

            // B frags straight from L2-resident bf16 w: lane = N column, elements = K (table order)
            BF16Frag bf[4];
#pragma unroll
            for (int nt = 0; nt < 4; ++nt) {
                for (int j = 0; j < 8; ++j) {
                    const unsigned short* p = bro + (size_t)(j < 4 ? 2 * j : 8 + 2 * j) * 4096 + nt * 16;
                    bf[nt].u[j] = (unsigned int)p[0] | ((unsigned int)p[4096] << 16);
                }
            }
            if (kb < 224) __builtin_prefetch(bro + bstep, 0, 1);
            bro += bstep;
#pragma unroll
            for (int nt = 0; nt < 4; ++nt) {
                acc[nt]     = wmma_bf16(a0.v, bf[nt].v, acc[nt]);
                acc[4 + nt] = wmma_bf16(a1.v, bf[nt].v, acc[4 + nt]);
            }
        }

        // e = exp(q), in the accumulators
#pragma unroll
        for (int i = 0; i < 8; ++i)
#pragma unroll
            for (int r = 0; r < 8; ++r) acc[i][r] = __expf(acc[i][r]);

        __syncthreads();                       // prior tile's S reads done
#pragma unroll
        for (int i = 0; i < 8; ++i) S[tid + 256 * i] = 0.0f;
        __syncthreads();

        // denominator: sum over batch (the 8 waves) per (v, col)
#pragma unroll
        for (int i = 0; i < 8; ++i) {
            const int rt = i >> 2, nt = i & 3;
#pragma unroll
            for (int r = 0; r < 8; ++r) {
                int vrow = rt * 16 + r + sel * 8;
                atomicAdd(&S[vrow * 64 + nt * 16 + lo], acc[i][r]);
            }
        }
        __syncthreads();

        // u[b,h,c] = sum_v (e/S) * v[b,v,c]
        float part[4] = {0.f, 0.f, 0.f, 0.f};
#pragma unroll
        for (int i = 0; i < 8; ++i) {
            const int rt = i >> 2, nt = i & 3;
            const int col = nt * 16 + lo;
#pragma unroll
            for (int r = 0; r < 8; ++r) {
                int vrow = rt * 16 + r + sel * 8;
                float beta = acc[i][r] / S[vrow * 64 + col];
                float vval = vin[(size_t)(b * 32 + vrow) * 4096 + cbase + col];
                part[nt] += beta * vval;
            }
        }
#pragma unroll
        for (int nt = 0; nt < 4; ++nt) {
            float tot = part[nt] + __shfl_xor(part[nt], 16, 32);  // merge row halves
            if (sel == 0)
                out[(size_t)(b * 32 + h) * 4096 + cbase + nt * 16 + lo] = tot;
        }
    }
}

// ---------------------------------------------------------------------------
extern "C" void kernel_launch(void* const* d_in, const int* in_sizes, int n_in,
                              void* d_out, int out_size, void* d_ws, size_t ws_size,
                              hipStream_t stream) {
    (void)in_sizes; (void)n_in; (void)out_size; (void)ws_size;
    const float* h    = (const float*)d_in[0];   // (8,32,256)
    const float* v    = (const float*)d_in[1];   // (8,32,4096)
    const float* W    = (const float*)d_in[2];   // (4096,256)
    const float* U    = (const float*)d_in[3];   // (256,256)
    const float* bias = (const float*)d_in[4];   // (256,)
    const float* w    = (const float*)d_in[5];   // (256,4096)
    float* out = (float*)d_out;                  // (8,32,4096)

    // Workspace layout (~6.9 MB):
    //   Uh 256KB | Wv 256KB | w_bf 2MB | v_bf 2MB | W_bf 2MB | h_bf 128KB | U_bf 128KB
    float*        Uh  = (float*)d_ws;
    float*        Wv  = Uh + 65536;
    unsigned int* wbf = (unsigned int*)(Wv + 65536);          // 524288 uints
    unsigned int* vbf = wbf + 524288;
    unsigned int* Wbf = vbf + 524288;
    unsigned int* hbf = Wbf + 524288;
    unsigned int* Ubf = hbf + 32768;

    // fp32 -> bf16, one pass each (memory bound, ~19.7 MB total traffic)
    cvt_f32_to_bf16<<<2048, 256, 0, stream>>>(w, wbf, 524288);
    cvt_f32_to_bf16<<<2048, 256, 0, stream>>>(v, vbf, 524288);
    cvt_f32_to_bf16<<<2048, 256, 0, stream>>>(W, Wbf, 524288);
    cvt_f32_to_bf16<<<128,  256, 0, stream>>>(h, hbf, 32768);
    cvt_f32_to_bf16<<<128,  256, 0, stream>>>(U, Ubf, 32768);

    // U_h = h@U + b  (M=256,N=256,K=256) ; W_v = v@W (M=256,N=256,K=4096)
    gemm_bf16_wmma<<<8, 256, 0, stream>>>((const unsigned short*)hbf,
                                          (const unsigned short*)Ubf,
                                          bias, Uh, 256, 256, 256);
    gemm_bf16_wmma<<<8, 256, 0, stream>>>((const unsigned short*)vbf,
                                          (const unsigned short*)Wbf,
                                          nullptr, Wv, 256, 256, 4096);

    // Fused tanh -> GEMM -> exp -> batch-norm -> v-reduction
    size_t smem = (size_t)256 * ASTR * 2 + 32 * 64 * 4;   // 143,360 B (< 320 KB WGP)
    fused_attention<<<dim3(32, 8), 256, smem, stream>>>(
        v, Uh, Wv, (const unsigned short*)wbf, out);
}
